// SemanticEncoder_22385369547374
// MI455X (gfx1250) — matrix operand order repaired and checked by reading, not compiled, not run on hardware
//
#include <hip/hip_runtime.h>

// ---------------------------------------------------------------------------
// SemanticEncoder block for MI455X (gfx1250, wave32, WMMA + async LDS DMA).
// Shapes fixed: B=4, S=2048, C=384, H=2, K=192, W=4, FF=1024.
// ---------------------------------------------------------------------------

typedef __attribute__((ext_vector_type(16))) _Float16 v16h;
typedef __attribute__((ext_vector_type(8)))  _Float16 v8h;
typedef __attribute__((ext_vector_type(8)))  float    v8f;

#define B_   4
#define S_   2048
#define C_   384
#define H_   2
#define K_   192
#define FF_  1024
#define FH_  512
#define R_   (B_ * S_)     // 8192 token rows
#define BH_  (B_ * H_)     // 8 (batch*head)
#define EPS_ 1e-6f
#define QSCALE_ 0.07216878364870323f   // 1/sqrt(192)

// --- helpers ---------------------------------------------------------------

__device__ __forceinline__ v16h ld_frag(const _Float16* p0, const _Float16* p1) {
  v8h lo = *(const v8h*)p0;
  v8h hi = *(const v8h*)p1;
  v16h r;
#pragma unroll
  for (int i = 0; i < 8; ++i) { r[i] = lo[i]; r[8 + i] = hi[i]; }
  return r;
}

// Async 16B global -> LDS copy (CDNA5 Tensor/async path, ASYNCcnt tracked).
// VDST = LDS byte address VGPR, VADDR = 64-bit global address VGPR pair.
__device__ __forceinline__ void async_copy16(void* lds_ptr, const void* gptr) {
  unsigned lds_off = (unsigned)(size_t)lds_ptr;
  asm volatile("global_load_async_to_lds_b128 %0, %1, off"
               :: "v"(lds_off), "v"(gptr)
               : "memory");
}
__device__ __forceinline__ void async_wait0() {
  asm volatile("s_wait_asynccnt 0" ::: "memory");
}

__device__ __forceinline__ float wave_sum32(float v) {
#pragma unroll
  for (int m = 16; m >= 1; m >>= 1) v += __shfl_xor(v, m, 32);
  return v;
}

// reduce across the 16 lanes of one half-wave (masks 1..8 keep bit4 fixed)
__device__ __forceinline__ float half_sum16(float v) {
#pragma unroll
  for (int m = 8; m >= 1; m >>= 1) v += __shfl_xor(v, m, 32);
  return v;
}
__device__ __forceinline__ float half_max16(float v) {
#pragma unroll
  for (int m = 8; m >= 1; m >>= 1) v = fmaxf(v, __shfl_xor(v, m, 32));
  return v;
}

// --- kernel 1/6: (optional residual add) + RMSNorm -> f16 ------------------
// one wave per row, C=384 -> 12 elements per lane.
__global__ void __launch_bounds__(256)
resid_rmsnorm_f16_kernel(const float* __restrict__ x,
                         const float* __restrict__ resid,   // may be null
                         const float* __restrict__ w,
                         float* __restrict__ x2_out,        // may be null
                         _Float16* __restrict__ nxh) {
  int wv = threadIdx.x >> 5, lane = threadIdx.x & 31;
  int row = blockIdx.x * 8 + wv;
  const float* xr = x + (size_t)row * C_;
  const float* rr = resid ? resid + (size_t)row * C_ : nullptr;
  float vals[12];
  float ss = 0.f;
#pragma unroll
  for (int i = 0; i < 12; ++i) {
    int c = lane + 32 * i;
    float v = xr[c] + (rr ? rr[c] : 0.f);
    vals[i] = v;
    ss += v * v;
  }
  ss = wave_sum32(ss);
  float rs = rsqrtf(ss * (1.0f / C_) + EPS_);
#pragma unroll
  for (int i = 0; i < 12; ++i) {
    int c = lane + 32 * i;
    if (x2_out) x2_out[(size_t)row * C_ + c] = vals[i];
    nxh[(size_t)row * C_ + c] = (_Float16)(vals[i] * rs * w[c]);
  }
}

// --- shared WMMA GEMM: out[M,N] = A[M,Kd](f16) * Wt[N,Kd]^T + bias ---------
// block tile 128x64, BK=32, 8 waves (4x2), each wave 32x32 via 2x2 WMMA accs.
// A tile is staged with async global->LDS DMA; B tile converts f32->f16 in
// registers (v_cvt_pk_f16_f32) on the way to LDS.
// epi: 0 = store f32  | 1 = qkv scatter (q scaled, k, v-transposed, f16)
//      2 = residual + store f32
__global__ void __launch_bounds__(256)
gemm_f16_kernel(const _Float16* __restrict__ A, const float* __restrict__ Wt,
                const float* __restrict__ bias, int epi,
                float* __restrict__ Cout, const float* __restrict__ resid,
                _Float16* __restrict__ qh, _Float16* __restrict__ kh,
                _Float16* __restrict__ vth, int N, int Kd) {
  __shared__ _Float16 As[128 * 32];
  __shared__ _Float16 Bs[64 * 32];
  int tid = threadIdx.x;
  int lane = tid & 31;
  int wv = tid >> 5;
  int lhalf = lane >> 4, lcol = lane & 15;
  int wy = wv & 3, wx = wv >> 2;          // 4 waves in M, 2 in N
  int bm = blockIdx.x * 128, bn = blockIdx.y * 64;

  v8f acc[2][2];
#pragma unroll
  for (int a = 0; a < 2; ++a)
#pragma unroll
    for (int b = 0; b < 2; ++b)
#pragma unroll
      for (int j = 0; j < 8; ++j) acc[a][b][j] = 0.f;

  int nkt = Kd >> 5;
  int ar = tid >> 1, acg = (tid & 1) * 16;   // A stage: 16 halves / thread
  int br = tid >> 2, bcg = (tid & 3) * 8;    // B stage: 8 f32->f16 / thread
  for (int kt = 0; kt < nkt; ++kt) {
    __syncthreads();
    {  // A tile: async DMA, no VGPR round trip
      const _Float16* src = A + (size_t)(bm + ar) * Kd + kt * 32 + acg;
      _Float16* dst = &As[ar * 32 + acg];
      async_copy16(dst, src);
      async_copy16(dst + 8, src + 8);
      if (kt + 1 < nkt) __builtin_prefetch(src + 32, 0, 3);
    }
    {  // B tile: load f32, convert, store f16
      const float* src = Wt + (size_t)(bn + br) * Kd + kt * 32 + bcg;
      _Float16* dst = &Bs[br * 32 + bcg];
#pragma unroll
      for (int i = 0; i < 8; ++i) dst[i] = (_Float16)src[i];
      if (kt + 1 < nkt) __builtin_prefetch(src + 32, 0, 3);
    }
    async_wait0();
    __syncthreads();

    v16h am[2], bf[2];
#pragma unroll
    for (int i = 0; i < 2; ++i) {
      const _Float16* p = &As[(wy * 32 + i * 16 + lcol) * 32 + lhalf * 8];
      am[i] = ld_frag(p, p + 16);
    }
#pragma unroll
    for (int i = 0; i < 2; ++i) {
      const _Float16* p = &Bs[(wx * 32 + i * 16 + lcol) * 32 + lhalf * 8];
      bf[i] = ld_frag(p, p + 16);
    }
#pragma unroll
    for (int mi = 0; mi < 2; ++mi)
#pragma unroll
      for (int ni = 0; ni < 2; ++ni)
        acc[mi][ni] = __builtin_amdgcn_wmma_f32_16x16x32_f16(
            false, am[mi], false, bf[ni], (short)0, acc[mi][ni], false, false);
  }

  // epilogue: element (row j+8*lhalf, col lcol) of each 16x16 tile
#pragma unroll
  for (int mi = 0; mi < 2; ++mi)
#pragma unroll
    for (int ni = 0; ni < 2; ++ni)
#pragma unroll
      for (int j = 0; j < 8; ++j) {
        int m = bm + wy * 32 + mi * 16 + j + 8 * lhalf;
        int n = bn + wx * 32 + ni * 16 + lcol;
        float val = acc[mi][ni][j] + bias[n];
        if (epi == 0) {
          Cout[(size_t)m * N + n] = val;
        } else if (epi == 2) {
          Cout[(size_t)m * N + n] = resid[(size_t)m * N + n] + val;
        } else {  // qkv scatter
          int sect = n / C_;          // 0=q 1=k 2=v
          int nn = n - sect * C_;
          int hh = nn / K_;
          int d  = nn - hh * K_;
          int bb = m >> 11;           // /S_
          int s  = m & (S_ - 1);
          size_t bhs = ((size_t)(bb * H_ + hh) * S_ + s);
          if (sect == 0)       qh[bhs * K_ + d] = (_Float16)(val * QSCALE_);
          else if (sect == 1)  kh[bhs * K_ + d] = (_Float16)val;
          else vth[((size_t)(bb * H_ + hh) * K_ + d) * S_ + s] = (_Float16)val;
        }
      }
}

// --- kernel 3: qe[bh,s,j] = qs[bh,s,:] . emb_rel_k[j,:]  (j=0..8) ----------
__global__ void __launch_bounds__(256)
qe_kernel(const _Float16* __restrict__ qh, const float* __restrict__ embk,
          float* __restrict__ qe) {
  int wv = threadIdx.x >> 5, lane = threadIdx.x & 31;
  int row = blockIdx.x * 8 + wv;            // bh*S + s
  const _Float16* qr = qh + (size_t)row * K_;
  float qv[6];
#pragma unroll
  for (int i = 0; i < 6; ++i) qv[i] = (float)qr[lane + 32 * i];
#pragma unroll
  for (int j = 0; j < 9; ++j) {
    float p = 0.f;
#pragma unroll
    for (int i = 0; i < 6; ++i) p += qv[i] * embk[j * K_ + lane + 32 * i];
    p = wave_sum32(p);
    if (lane == 0) qe[(size_t)row * 9 + j] = p;
  }
}

// --- kernel 4: flash attention (streams K/V, never spills scores) ----------
// grid (S/64, B*H), block 128 = 4 waves; wave owns 16 q rows, full head dim.
// K/V tiles staged via async global->LDS DMA (ASYNCcnt).
__global__ void __launch_bounds__(128)
flash_attn_kernel(const _Float16* __restrict__ qh, const _Float16* __restrict__ kh,
                  const _Float16* __restrict__ vth, const float* __restrict__ qe,
                  float* __restrict__ attn, float* __restrict__ mstat,
                  float* __restrict__ lstat) {
  __shared__ _Float16 kt[32 * K_];        // K tile: [key][dim]
  __shared__ _Float16 vt[K_ * 32];        // V tile: [dim][key] (pre-transposed)
  __shared__ _Float16 plds[4][16 * 32];   // per-wave p relayout buffer
  int tid = threadIdx.x;
  int wv = tid >> 5, lane = tid & 31;
  int lhalf = lane >> 4, lcol = lane & 15;
  int bh = blockIdx.y;
  int b = bh >> 1, h = bh & 1;
  int qrow = blockIdx.x * 64 + wv * 16;

  // Q fragments for this wave's 16 rows (held for the whole loop)
  v16h qa[6];
  {
    const _Float16* qbase = qh + ((size_t)bh * S_ + qrow + lcol) * K_;
#pragma unroll
    for (int kc = 0; kc < 6; ++kc) {
      const _Float16* p = qbase + kc * 32 + lhalf * 8;
      qa[kc] = ld_frag(p, p + 16);
    }
  }

  float mrow[8], lrow[8];
  v8f o[12];
#pragma unroll
  for (int j = 0; j < 8; ++j) { mrow[j] = -1e30f; lrow[j] = 0.f; }
#pragma unroll
  for (int t = 0; t < 12; ++t)
#pragma unroll
    for (int j = 0; j < 8; ++j) o[t][j] = 0.f;

  int kr_ = tid >> 2, kcg = (tid & 3) * 48;   // K staging coords
  for (int kb = 0; kb < S_; kb += 32) {
    __syncthreads();
    {  // stage K tile: 32 rows x 192 halves via async DMA (96B / thread)
      const _Float16* src = kh + ((size_t)bh * S_ + kb + kr_) * K_ + kcg;
      _Float16* dst = &kt[kr_ * K_ + kcg];
#pragma unroll
      for (int i = 0; i < 6; ++i) async_copy16(dst + i * 8, src + i * 8);
    }
    {  // stage Vt tile: 192 rows x 32 halves via async DMA (64B / row)
#pragma unroll
      for (int rr = 0; rr < 2; ++rr) {
        int r = tid + rr * 128;
        if (r < K_) {
          const _Float16* src = vth + ((size_t)bh * K_ + r) * S_ + kb;
          _Float16* dst = &vt[r * 32];
#pragma unroll
          for (int i = 0; i < 4; ++i) async_copy16(dst + i * 8, src + i * 8);
        }
      }
    }
    async_wait0();
    __syncthreads();

    // scores: two 16x16 tiles (keys kb..+15, kb+16..+31), K reduced over 192
    v8f s0, s1;
#pragma unroll
    for (int j = 0; j < 8; ++j) { s0[j] = 0.f; s1[j] = 0.f; }
#pragma unroll
    for (int kc = 0; kc < 6; ++kc) {
      const _Float16* p0 = &kt[lcol * K_ + kc * 32 + lhalf * 8];
      const _Float16* p1 = &kt[(16 + lcol) * K_ + kc * 32 + lhalf * 8];
      v16h b0 = ld_frag(p0, p0 + 16);
      v16h b1 = ld_frag(p1, p1 + 16);
      s0 = __builtin_amdgcn_wmma_f32_16x16x32_f16(false, qa[kc], false, b0,
                                                  (short)0, s0, false, false);
      s1 = __builtin_amdgcn_wmma_f32_16x16x32_f16(false, qa[kc], false, b1,
                                                  (short)0, s1, false, false);
    }

    // banded relative-K logits: scores[q,kk] += qe[q, kk-q+4] for |kk-q|<=4
#pragma unroll
    for (int j = 0; j < 8; ++j) {
      int m = qrow + j + 8 * lhalf;
      int d0 = (kb + lcol) - m;
      if (d0 >= -4 && d0 <= 4) s0[j] += qe[((size_t)bh * S_ + m) * 9 + d0 + 4];
      int d1 = d0 + 16;
      if (d1 >= -4 && d1 <= 4) s1[j] += qe[((size_t)bh * S_ + m) * 9 + d1 + 4];
    }

    // online softmax per row (rows live in half-waves; reduce over 16 lanes)
    float sc[8], p0f[8], p1f[8];
#pragma unroll
    for (int j = 0; j < 8; ++j) {
      float tm = half_max16(fmaxf(s0[j], s1[j]));
      float mn = fmaxf(mrow[j], tm);
      sc[j] = __expf(mrow[j] - mn);
      p0f[j] = __expf(s0[j] - mn);
      p1f[j] = __expf(s1[j] - mn);
      float r = half_sum16(p0f[j] + p1f[j]);
      lrow[j] = lrow[j] * sc[j] + r;
      mrow[j] = mn;
    }
#pragma unroll
    for (int t = 0; t < 12; ++t)
#pragma unroll
      for (int j = 0; j < 8; ++j) o[t][j] *= sc[j];

    // relayout p (C layout) -> A fragment layout through per-wave LDS
    _Float16* pw = plds[wv];
#pragma unroll
    for (int j = 0; j < 8; ++j) {
      int r = j + 8 * lhalf;
      pw[r * 32 + lcol]      = (_Float16)p0f[j];
      pw[r * 32 + 16 + lcol] = (_Float16)p1f[j];
    }
    asm volatile("s_wait_dscnt 0" ::: "memory");
    const _Float16* pp = &pw[lcol * 32 + lhalf * 8];
    v16h pa = ld_frag(pp, pp + 16);

    // out[16,192] += p[16,32] x v[32,192]
#pragma unroll
    for (int nd = 0; nd < 12; ++nd) {
      const _Float16* vp = &vt[(nd * 16 + lcol) * 32 + lhalf * 8];
      v16h bv = ld_frag(vp, vp + 16);
      o[nd] = __builtin_amdgcn_wmma_f32_16x16x32_f16(false, pa, false, bv,
                                                     (short)0, o[nd], false, false);
    }
  }

  // epilogue: normalize and scatter into [B,S,C] (heads interleaved)
#pragma unroll
  for (int j = 0; j < 8; ++j) {
    int m = qrow + j + 8 * lhalf;
    float inv = 1.0f / lrow[j];
    float* dst = attn + ((size_t)b * S_ + m) * C_ + h * K_;
#pragma unroll
    for (int nd = 0; nd < 12; ++nd) dst[nd * 16 + lcol] = o[nd][j] * inv;
  }
  if (lcol == 0) {
#pragma unroll
    for (int j = 0; j < 8; ++j) {
      int m = qrow + j + 8 * lhalf;
      mstat[(size_t)bh * S_ + m] = mrow[j];
      lstat[(size_t)bh * S_ + m] = lrow[j];
    }
  }
}

// --- kernel 5: banded rel_v term: out[q] += sum_d p[q,q+d] * emb_rel_v[d+4]
// recompute the 9 band probabilities from saved (m,l) stats; one wave per row.
__global__ void __launch_bounds__(256)
band_relv_kernel(const _Float16* __restrict__ qh, const _Float16* __restrict__ kh,
                 const float* __restrict__ qe, const float* __restrict__ embv,
                 const float* __restrict__ mstat, const float* __restrict__ lstat,
                 float* __restrict__ attn) {
  int wv = threadIdx.x >> 5, lane = threadIdx.x & 31;
  int row = blockIdx.x * 8 + wv;       // bh*S + q
  int bh = row >> 11;
  int q = row & (S_ - 1);
  int b = bh >> 1, h = bh & 1;
  const _Float16* qr = qh + (size_t)row * K_;
  float qv[6];
#pragma unroll
  for (int i = 0; i < 6; ++i) qv[i] = (float)qr[lane + 32 * i];
  float m = mstat[row];
  float linv = 1.0f / lstat[row];
  float acc[6] = {0.f, 0.f, 0.f, 0.f, 0.f, 0.f};
#pragma unroll
  for (int d = -4; d <= 4; ++d) {
    int kk = q + d;
    if ((unsigned)kk >= (unsigned)S_) continue;   // uniform per wave
    const _Float16* kr = kh + ((size_t)bh * S_ + kk) * K_;
    float s = 0.f;
#pragma unroll
    for (int i = 0; i < 6; ++i) s += qv[i] * (float)kr[lane + 32 * i];
    s = wave_sum32(s) + qe[(size_t)row * 9 + d + 4];
    float p = __expf(s - m) * linv;
#pragma unroll
    for (int i = 0; i < 6; ++i)
      acc[i] += p * embv[(d + 4) * K_ + lane + 32 * i];
  }
  float* dst = attn + ((size_t)b * S_ + q) * C_ + h * K_;
#pragma unroll
  for (int i = 0; i < 6; ++i) dst[lane + 32 * i] += acc[i];
}

// --- kernel 8: SwiGLU + RMSNorm -> f16 (one wave per row, FH=512) ----------
__global__ void __launch_bounds__(256)
swiglu_rmsnorm_kernel(const float* __restrict__ h1, const float* __restrict__ w,
                      _Float16* __restrict__ gh) {
  int wv = threadIdx.x >> 5, lane = threadIdx.x & 31;
  int row = blockIdx.x * 8 + wv;
  const float* hr = h1 + (size_t)row * FF_;
  float g[16];
  float ss = 0.f;
#pragma unroll
  for (int i = 0; i < 16; ++i) {
    int c = lane + 32 * i;
    float x1 = hr[c];
    float x2 = hr[FH_ + c];
    float v = x1 * (x2 / (1.0f + __expf(-x2)));   // x1 * silu(x2)
    g[i] = v;
    ss += v * v;
  }
  ss = wave_sum32(ss);
  float rs = rsqrtf(ss * (1.0f / FH_) + EPS_);
#pragma unroll
  for (int i = 0; i < 16; ++i) {
    int c = lane + 32 * i;
    gh[(size_t)row * FH_ + c] = (_Float16)(g[i] * rs * w[c]);
  }
}

// ---------------------------------------------------------------------------

extern "C" void kernel_launch(void* const* d_in, const int* in_sizes, int n_in,
                              void* d_out, int out_size, void* d_ws, size_t ws_size,
                              hipStream_t stream) {
  const float* x            = (const float*)d_in[0];
  const float* attn_norm_w  = (const float*)d_in[1];
  const float* qkv_w        = (const float*)d_in[2];
  const float* qkv_b        = (const float*)d_in[3];
  const float* emb_rel_k    = (const float*)d_in[4];   // [9,192]
  const float* emb_rel_v    = (const float*)d_in[5];   // [9,192]
  const float* mlp_norm_w   = (const float*)d_in[6];
  const float* fc1_w        = (const float*)d_in[7];
  const float* fc1_b        = (const float*)d_in[8];
  const float* swiglu_w     = (const float*)d_in[9];
  const float* fc2_w        = (const float*)d_in[10];
  const float* fc2_b        = (const float*)d_in[11];
  float* out = (float*)d_out;

  char* ws = (char*)d_ws;
  size_t off = 0;
  auto alloc = [&](size_t bytes) -> void* {
    off = (off + 255) & ~(size_t)255;
    void* p = ws + off;
    off += bytes;
    return p;
  };
  _Float16* nx_h  = (_Float16*)alloc((size_t)R_ * C_ * 2);
  _Float16* qh    = (_Float16*)alloc((size_t)BH_ * S_ * K_ * 2);
  _Float16* kh    = (_Float16*)alloc((size_t)BH_ * S_ * K_ * 2);
  _Float16* vth   = (_Float16*)alloc((size_t)BH_ * K_ * S_ * 2);
  float*    qe    = (float*)alloc((size_t)BH_ * S_ * 9 * 4);
  float*    mstat = (float*)alloc((size_t)BH_ * S_ * 4);
  float*    lstat = (float*)alloc((size_t)BH_ * S_ * 4);
  float*    attn  = (float*)alloc((size_t)R_ * C_ * 4);
  float*    x2    = (float*)alloc((size_t)R_ * C_ * 4);
  _Float16* nx2_h = (_Float16*)alloc((size_t)R_ * C_ * 2);
  float*    h1    = (float*)alloc((size_t)R_ * FF_ * 4);
  _Float16* g_h   = (_Float16*)alloc((size_t)R_ * FH_ * 2);
  (void)ws_size; (void)in_sizes; (void)n_in; (void)out_size;

  // 1. attn RMSNorm -> f16
  resid_rmsnorm_f16_kernel<<<R_ / 8, 256, 0, stream>>>(x, nullptr, attn_norm_w,
                                                       nullptr, nx_h);
  // 2. QKV GEMM [8192,384]x[1152,384]^T, scatter q(scaled)/k/v^T as f16
  gemm_f16_kernel<<<dim3(R_ / 128, (3 * C_) / 64), 256, 0, stream>>>(
      nx_h, qkv_w, qkv_b, /*epi=*/1, nullptr, nullptr, qh, kh, vth, 3 * C_, C_);
  // 3. banded relative-K logit table qe[bh,s,9]
  qe_kernel<<<(BH_ * S_) / 8, 256, 0, stream>>>(qh, emb_rel_k, qe);
  // 4. flash attention (scores + band + softmax + p.v), save (m,l) stats
  flash_attn_kernel<<<dim3(S_ / 64, BH_), 128, 0, stream>>>(qh, kh, vth, qe,
                                                            attn, mstat, lstat);
  // 5. banded rel_v correction: attn += sum_d p[q,q+d]*emb_rel_v[d+4]
  band_relv_kernel<<<(BH_ * S_) / 8, 256, 0, stream>>>(qh, kh, qe, emb_rel_v,
                                                       mstat, lstat, attn);
  // 6. x2 = x + attn ; mlp RMSNorm -> f16
  resid_rmsnorm_f16_kernel<<<R_ / 8, 256, 0, stream>>>(x, attn, mlp_norm_w,
                                                       x2, nx2_h);
  // 7. fc1 GEMM [8192,384]x[1024,384]^T -> h1 f32
  gemm_f16_kernel<<<dim3(R_ / 128, FF_ / 64), 256, 0, stream>>>(
      nx2_h, fc1_w, fc1_b, /*epi=*/0, h1, nullptr, nullptr, nullptr, nullptr,
      FF_, C_);
  // 8. SwiGLU + RMSNorm -> f16
  swiglu_rmsnorm_kernel<<<R_ / 8, 256, 0, stream>>>(h1, swiglu_w, g_h);
  // 9. fc2 GEMM [8192,512]x[384,512]^T + residual -> d_out
  gemm_f16_kernel<<<dim3(R_ / 128, C_ / 64), 256, 0, stream>>>(
      g_h, fc2_w, fc2_b, /*epi=*/2, out, x2, nullptr, nullptr, nullptr,
      C_, FH_);
}
